// TransformerDecoderBlock_609885356657
// MI455X (gfx1250) — matrix operand order repaired and checked by reading, not compiled, not run on hardware
//
#include <hip/hip_runtime.h>
#include <hip/hip_bf16.h>
#include <math.h>
#include <stdint.h>

// ---------------------------------------------------------------------------
// Problem constants (fixed by the reference).
// ---------------------------------------------------------------------------
constexpr int B_   = 2;
constexpr int S_   = 2048;
constexpr int D_   = 2048;
constexpr int H_   = 16;
constexpr int KVH_ = 4;
constexpr int HD_  = 128;
constexpr int F_   = 8192;
constexpr float EPS_ = 1e-6f;

typedef _Float16 v16h  __attribute__((ext_vector_type(16)));
typedef _Float16 half8 __attribute__((ext_vector_type(8)));
typedef _Float16 half4 __attribute__((ext_vector_type(4)));
typedef float    v8f   __attribute__((ext_vector_type(8)));

// D = A(16x32) * B(32x16) + C, f32 accumulate  -> v_wmma_f32_16x16x32_f16
__device__ inline v8f wmma_f16(v16h a, v16h b, v8f c) {
  return __builtin_amdgcn_wmma_f32_16x16x32_f16(
      /*neg_a=*/false, a, /*neg_b=*/false, b,
      /*c_mod=*/(short)0, c, /*reuse_a=*/false, /*reuse_b=*/false);
}

// ---------------------------------------------------------------------------
// CDNA5 async global->LDS copy (ASYNCcnt path). The low 32 bits of a generic
// pointer to __shared__ are the LDS byte offset (ISA 10.2 aperture rules).
// Requires 16B alignment on both addresses.
// ---------------------------------------------------------------------------
__device__ inline void async_b128(const _Float16* g, _Float16* l) {
  unsigned lds_off = (unsigned)(uintptr_t)l;
  asm volatile("global_load_async_to_lds_b128 %0, %1, off"
               :: "v"(lds_off), "v"(g) : "memory");
}
__device__ inline void wait_async0() {
  asm volatile("s_wait_asynccnt 0x0" ::: "memory");
}

// ---------------------------------------------------------------------------
// f32 -> f16 cast, vectorized 4-wide (all weight sizes are multiples of 4).
// ---------------------------------------------------------------------------
__global__ void f32_to_f16_kernel(const float* __restrict__ src,
                                  _Float16* __restrict__ dst, int n4) {
  int i = blockIdx.x * blockDim.x + threadIdx.x;
  if (i < n4) {
    const float4 v = ((const float4*)src)[i];
    half4 h;
    h[0] = (_Float16)v.x; h[1] = (_Float16)v.y;
    h[2] = (_Float16)v.z; h[3] = (_Float16)v.w;
    ((half4*)dst)[i] = h;
  }
}

// ---------------------------------------------------------------------------
// RMSNorm: one block (256 threads = 8 waves) per row of D_ elements, f16 out.
// ---------------------------------------------------------------------------
__global__ __launch_bounds__(256) void rmsnorm_f16_kernel(
    const float* __restrict__ x, const float* __restrict__ g,
    _Float16* __restrict__ out) {
  __shared__ float red[8];
  const int row = blockIdx.x;
  const float* xr = x + (size_t)row * D_;
  float ss = 0.f;
  for (int i = threadIdx.x; i < D_; i += 256) { float v = xr[i]; ss += v * v; }
  ss += __shfl_xor(ss, 16, 32);
  ss += __shfl_xor(ss, 8, 32);
  ss += __shfl_xor(ss, 4, 32);
  ss += __shfl_xor(ss, 2, 32);
  ss += __shfl_xor(ss, 1, 32);
  if ((threadIdx.x & 31) == 0) red[threadIdx.x >> 5] = ss;
  __syncthreads();
  float tot = 0.f;
#pragma unroll
  for (int w = 0; w < 8; ++w) tot += red[w];
  const float sc = rsqrtf(tot / (float)D_ + EPS_);
  _Float16* outr = out + (size_t)row * D_;
  for (int i = threadIdx.x; i < D_; i += 256)
    outr[i] = (_Float16)(xr[i] * sc * g[i]);
}

// ---------------------------------------------------------------------------
// RoPE in-place on an f16 (B,S,heads,HD) buffer. One thread per (d, d+64)
// pair; cos/sin are (S, HD) with the two halves duplicated.
// ---------------------------------------------------------------------------
__global__ void rope_f16_kernel(_Float16* __restrict__ t,
                                const float* __restrict__ cosb,
                                const float* __restrict__ sinb,
                                int heads, int total_pairs) {
  int idx = blockIdx.x * blockDim.x + threadIdx.x;
  if (idx >= total_pairs) return;
  const int d = idx & 63;              // 0..HD/2-1
  const int r = idx >> 6;              // (b*S + s)*heads + head
  const int s = (r / heads) % S_;
  const size_t base = (size_t)r * HD_;
  const float c  = cosb[(size_t)s * HD_ + d];
  const float sn = sinb[(size_t)s * HD_ + d];
  const float a  = (float)t[base + d];
  const float b2 = (float)t[base + d + 64];
  t[base + d]      = (_Float16)(a * c - b2 * sn);
  t[base + d + 64] = (_Float16)(b2 * c + a * sn);
}

// ---------------------------------------------------------------------------
// WMMA GEMM: C[M,N] = A[M,K] (f16 row-major) x Bw[K,N] (f16 row-major).
// Block tile 128x128x32, 8 waves in 4x2, wave tile 32x64 (2x4 WMMA frags).
// Double-buffered async global->LDS staging, one barrier per K tile.
// Templated fused epilogues.
// ---------------------------------------------------------------------------
enum Epi { EPI_F16 = 0, EPI_SILU_F16 = 1, EPI_MUL_F16 = 2, EPI_RESID_F32 = 3 };

template <int EPI>
__global__ __launch_bounds__(256) void gemm_wmma_kernel(
    const _Float16* __restrict__ A, const _Float16* __restrict__ Bw,
    _Float16* __restrict__ outh, float* __restrict__ outf,
    const float* __restrict__ resid, const _Float16* __restrict__ other,
    int M, int N, int K) {
  __shared__ _Float16 As[2][128][40];   // +8 half pad: conflict-free reads
  __shared__ _Float16 Bs[2][32][136];

  const int tid  = threadIdx.x;
  const int lane = tid & 31;
  const int wave = tid >> 5;
  const int wm   = wave >> 1;        // 0..3
  const int wn   = wave & 1;         // 0..1
  const int row0 = blockIdx.y * 128;
  const int col0 = blockIdx.x * 128;
  const int lrow = lane & 15;
  const int hsel = lane >> 4;        // 0 or 1
  const int akoff = hsel * 8;
  const int bkoff = hsel * 16;

  v8f acc[2][4];
  for (int i = 0; i < 2; ++i)
    for (int j = 0; j < 4; ++j)
      for (int e = 0; e < 8; ++e) acc[i][j][e] = 0.f;

  const int ar = tid >> 1;           // 0..127
  const int ac = (tid & 1) * 16;     // 0 / 16
  const int br = tid >> 3;           // 0..31
  const int bc = (tid & 7) * 16;     // 0..112

  auto issue_tile = [&](int buf, int k0) {
    const _Float16* ag = A + (size_t)(row0 + ar) * K + k0 + ac;
    async_b128(ag,     &As[buf][ar][ac]);
    async_b128(ag + 8, &As[buf][ar][ac + 8]);
    const _Float16* bg = Bw + (size_t)(k0 + br) * N + col0 + bc;
    async_b128(bg,     &Bs[buf][br][bc]);
    async_b128(bg + 8, &Bs[buf][br][bc + 8]);
  };

  issue_tile(0, 0);
  int buf = 0;
  for (int k0 = 0; k0 < K; k0 += 32, buf ^= 1) {
    wait_async0();         // this wave's stage into As/Bs[buf] is in LDS
    __syncthreads();       // every wave's stage is in LDS; prev tile consumed
    if (k0 + 32 < K) issue_tile(buf ^ 1, k0 + 32);  // prefetch under compute

    v16h af[2];
#pragma unroll
    for (int fm = 0; fm < 2; ++fm) {
      const _Float16* ap = &As[buf][wm * 32 + fm * 16 + lrow][0];
      half8 lo = *(const half8*)(ap + akoff);
      half8 hi = *(const half8*)(ap + 16 + akoff);
#pragma unroll
      for (int e = 0; e < 8; ++e) { af[fm][e] = lo[e]; af[fm][8 + e] = hi[e]; }
    }
    v16h bf[4];
#pragma unroll
    for (int fn = 0; fn < 4; ++fn) {
      const int c = wn * 64 + fn * 16 + lrow;
#pragma unroll
      for (int e = 0; e < 16; ++e) bf[fn][e] = Bs[buf][akoff + (e < 8 ? e : e + 8)][c];
    }
#pragma unroll
    for (int fm = 0; fm < 2; ++fm) {
#pragma unroll
      for (int fn = 0; fn < 4; ++fn) {
        acc[fm][fn] = wmma_f16(af[fm], bf[fn], acc[fm][fn]);
      }
    }
  }

  const int orow = hsel * 8;
#pragma unroll
  for (int fm = 0; fm < 2; ++fm) {
#pragma unroll
    for (int fn = 0; fn < 4; ++fn) {
#pragma unroll
      for (int e = 0; e < 8; ++e) {
        const int r = row0 + wm * 32 + fm * 16 + orow + e;
        const int c = col0 + wn * 64 + fn * 16 + lrow;
        const size_t idx = (size_t)r * N + c;
        const float v = acc[fm][fn][e];
        if constexpr (EPI == EPI_F16) {
          outh[idx] = (_Float16)v;
        } else if constexpr (EPI == EPI_SILU_F16) {
          outh[idx] = (_Float16)(v / (1.f + __expf(-v)));
        } else if constexpr (EPI == EPI_MUL_F16) {
          outh[idx] = (_Float16)(v * (float)other[idx]);
        } else {
          outf[idx] = v + resid[idx];
        }
      }
    }
  }
}

// ---------------------------------------------------------------------------
// Flash attention (causal, GQA). Block = (b, h, 128-row q tile), 8 waves,
// wave handles 16 q rows. Keys streamed in blocks of 32 with double-buffered
// async K/V staging (one barrier per key block); online softmax; P goes
// through LDS (same-wave DS ordering, no barrier) to switch from D-fragment
// layout to A-fragment layout for the PV WMMA.
// ---------------------------------------------------------------------------
__global__ __launch_bounds__(256) void attn_flash_kernel(
    const _Float16* __restrict__ Q, const _Float16* __restrict__ Kc,
    const _Float16* __restrict__ Vc, _Float16* __restrict__ Oout) {
  __shared__ _Float16 Kt[2][32][136];
  __shared__ _Float16 Vt[2][32][136];
  __shared__ _Float16 Pt[8][16][40];

  const int b = blockIdx.z, h = blockIdx.y, qt = blockIdx.x;
  const int kvh  = h / (H_ / KVH_);
  const int tid  = threadIdx.x;
  const int lane = tid & 31;
  const int wave = tid >> 5;
  const int lrow = lane & 15;
  const int hsel = lane >> 4;
  const int qrow0 = qt * 128 + wave * 16;
  const float scale = 0.08838834764831845f;  // 1/sqrt(128)
  const size_t qld  = (size_t)H_ * HD_;      // 2048
  const size_t kvld = (size_t)KVH_ * HD_;    // 512

  // Q A-fragments (16 rows x HD=128 -> four 16x32 K-chunks), kept in VGPRs.
  v16h qa[4];
  {
    const _Float16* qp =
        Q + (size_t)(b * S_ + qrow0 + lrow) * qld + (size_t)h * HD_;
    const int akoff = hsel * 16;
#pragma unroll
    for (int kc = 0; kc < 4; ++kc) {
      const _Float16* p = qp + kc * 32;
      half8 lo = *(const half8*)(p + akoff);
      half8 hi = *(const half8*)(p + 8 + akoff);
#pragma unroll
      for (int e = 0; e < 8; ++e) { qa[kc][e] = lo[e]; qa[kc][8 + e] = hi[e]; }
    }
  }

  v8f o[8];
  for (int n = 0; n < 8; ++n)
    for (int e = 0; e < 8; ++e) o[n][e] = 0.f;
  float mrow[8], lsum[8];
  for (int e = 0; e < 8; ++e) { mrow[e] = -3.0e38f; lsum[e] = 0.f; }

  const int nkb = (qt + 1) * 4;      // key blocks of 32, causal extent
  const int cr = tid >> 3;           // 0..31
  const int cc = (tid & 7) * 16;     // 0..112

  auto issue_kv = [&](int kbuf, int key0) {
    const size_t rowoff =
        ((size_t)(b * S_) + key0 + cr) * kvld + (size_t)kvh * HD_ + cc;
    async_b128(Kc + rowoff,     &Kt[kbuf][cr][cc]);
    async_b128(Kc + rowoff + 8, &Kt[kbuf][cr][cc + 8]);
    async_b128(Vc + rowoff,     &Vt[kbuf][cr][cc]);
    async_b128(Vc + rowoff + 8, &Vt[kbuf][cr][cc + 8]);
  };

  issue_kv(0, 0);
  int buf = 0;
  for (int kb = 0; kb < nkb; ++kb, buf ^= 1) {
    const int key0 = kb * 32;
    wait_async0();
    __syncthreads();
    if (kb + 1 < nkb) issue_kv(buf ^ 1, key0 + 32);  // prefetch under compute

    // scores S = Q * K^T for 32 keys (two 16-key fragments)
    v8f s0, s1;
    for (int e = 0; e < 8; ++e) { s0[e] = 0.f; s1[e] = 0.f; }
    const int bkoff = hsel * 16;
#pragma unroll
    for (int kc = 0; kc < 4; ++kc) {
      half8 lo0 = *(const half8*)&Kt[buf][lrow][kc * 32 + bkoff];
      half8 hi0 = *(const half8*)&Kt[buf][lrow][kc * 32 + bkoff + 8];
      half8 lo1 = *(const half8*)&Kt[buf][16 + lrow][kc * 32 + bkoff];
      half8 hi1 = *(const half8*)&Kt[buf][16 + lrow][kc * 32 + bkoff + 8];
      v16h b0, b1;
#pragma unroll
      for (int e = 0; e < 8; ++e) {
        b0[e] = lo0[e]; b0[8 + e] = hi0[e];
        b1[e] = lo1[e]; b1[8 + e] = hi1[e];
      }
      s0 = wmma_f16(qa[kc], b0, s0);
      s1 = wmma_f16(qa[kc], b1, s1);
    }

    // scale + causal mask + online softmax update
    const int rbase = qrow0 + hsel * 8;
    float pmax[8];
#pragma unroll
    for (int e = 0; e < 8; ++e) {
      s0[e] *= scale; s1[e] *= scale;
      const int r = rbase + e;
      if (key0 + lrow > r)      s0[e] = -1.0e30f;
      if (key0 + 16 + lrow > r) s1[e] = -1.0e30f;
      float v = fmaxf(s0[e], s1[e]);
      v = fmaxf(v, __shfl_xor(v, 1, 32));
      v = fmaxf(v, __shfl_xor(v, 2, 32));
      v = fmaxf(v, __shfl_xor(v, 4, 32));
      v = fmaxf(v, __shfl_xor(v, 8, 32));
      pmax[e] = v;
    }
    float alpha[8];
#pragma unroll
    for (int e = 0; e < 8; ++e) {
      const float mnew = fmaxf(mrow[e], pmax[e]);
      alpha[e] = __expf(mrow[e] - mnew);
      mrow[e] = mnew;
      s0[e] = __expf(s0[e] - mnew);
      s1[e] = __expf(s1[e] - mnew);
      float rs = s0[e] + s1[e];
      rs += __shfl_xor(rs, 1, 32);
      rs += __shfl_xor(rs, 2, 32);
      rs += __shfl_xor(rs, 4, 32);
      rs += __shfl_xor(rs, 8, 32);
      lsum[e] = lsum[e] * alpha[e] + rs;
    }
#pragma unroll
    for (int n = 0; n < 8; ++n) {
#pragma unroll
      for (int e = 0; e < 8; ++e) o[n][e] *= alpha[e];
    }

    // D-layout P -> LDS -> A-layout for PV. Same-wave DS ops are in order
    // and the region is wave-private, so no barrier is needed here.
#pragma unroll
    for (int e = 0; e < 8; ++e) {
      Pt[wave][hsel * 8 + e][lrow]      = (_Float16)s0[e];
      Pt[wave][hsel * 8 + e][16 + lrow] = (_Float16)s1[e];
    }
    v16h pa;
    {
      const int akoff = hsel * 8;
      const _Float16* pp = &Pt[wave][lrow][0];
      half8 lo = *(const half8*)(pp + akoff);
      half8 hi = *(const half8*)(pp + 16 + akoff);
#pragma unroll
      for (int e = 0; e < 8; ++e) { pa[e] = lo[e]; pa[8 + e] = hi[e]; }
    }
#pragma unroll
    for (int n = 0; n < 8; ++n) {
      v16h bv;
#pragma unroll
      for (int e = 0; e < 16; ++e)
        bv[e] = Vt[buf][hsel * 8 + (e < 8 ? e : e + 8)][n * 16 + lrow];
      o[n] = wmma_f16(pa, bv, o[n]);
    }
  }

  // normalize and write (b, s, h*HD + d) f16
#pragma unroll
  for (int n = 0; n < 8; ++n) {
#pragma unroll
    for (int e = 0; e < 8; ++e) {
      const int r = qrow0 + hsel * 8 + e;
      const float v = o[n][e] / lsum[e];
      Oout[(size_t)(b * S_ + r) * qld + (size_t)h * HD_ + n * 16 + lrow] =
          (_Float16)v;
    }
  }
}

// ---------------------------------------------------------------------------
// Host-side orchestration.
// ---------------------------------------------------------------------------
extern "C" void kernel_launch(void* const* d_in, const int* in_sizes, int n_in,
                              void* d_out, int out_size, void* d_ws,
                              size_t ws_size, hipStream_t stream) {
  (void)in_sizes; (void)n_in; (void)out_size; (void)ws_size;
  const float* x      = (const float*)d_in[0];
  const float* g_in   = (const float*)d_in[1];
  const float* g_out  = (const float*)d_in[2];
  const float* w_q    = (const float*)d_in[3];
  const float* w_k    = (const float*)d_in[4];
  const float* w_v    = (const float*)d_in[5];
  const float* w_o    = (const float*)d_in[6];
  const float* w_gate = (const float*)d_in[7];
  const float* w_up   = (const float*)d_in[8];
  const float* w_down = (const float*)d_in[9];
  const float* cosb   = (const float*)d_in[10];
  const float* sinb   = (const float*)d_in[11];
  float* out = (float*)d_out;

  char* p = (char*)d_ws;
  auto alloc = [&](size_t elems, size_t esz) -> void* {
    void* r = (void*)p;
    p += ((elems * esz) + 255) & ~(size_t)255;
    return r;
  };
  const size_t MS = (size_t)B_ * S_;          // 4096 rows
  _Float16* wq_h  = (_Float16*)alloc((size_t)D_ * D_, 2);
  _Float16* wk_h  = (_Float16*)alloc((size_t)D_ * KVH_ * HD_, 2);
  _Float16* wv_h  = (_Float16*)alloc((size_t)D_ * KVH_ * HD_, 2);
  _Float16* wo_h  = (_Float16*)alloc((size_t)D_ * D_, 2);
  _Float16* wg_h  = (_Float16*)alloc((size_t)D_ * F_, 2);
  _Float16* wu_h  = (_Float16*)alloc((size_t)D_ * F_, 2);
  _Float16* wd_h  = (_Float16*)alloc((size_t)F_ * D_, 2);
  _Float16* xn_h  = (_Float16*)alloc(MS * D_, 2);
  _Float16* q_h   = (_Float16*)alloc(MS * D_, 2);
  _Float16* k_h   = (_Float16*)alloc(MS * KVH_ * HD_, 2);
  _Float16* v_h   = (_Float16*)alloc(MS * KVH_ * HD_, 2);
  _Float16* at_h  = (_Float16*)alloc(MS * D_, 2);
  float*    x1    = (float*)alloc(MS * D_, 4);
  _Float16* xn2_h = (_Float16*)alloc(MS * D_, 2);
  _Float16* g_h   = (_Float16*)alloc(MS * F_, 2);
  _Float16* hh    = (_Float16*)alloc(MS * F_, 2);

  auto cvt = [&](const float* s, _Float16* d, size_t n) {
    const size_t n4 = n / 4;
    f32_to_f16_kernel<<<dim3((unsigned)((n4 + 255) / 256)), dim3(256), 0,
                        stream>>>(s, d, (int)n4);
  };
  cvt(w_q, wq_h, (size_t)D_ * D_);
  cvt(w_k, wk_h, (size_t)D_ * KVH_ * HD_);
  cvt(w_v, wv_h, (size_t)D_ * KVH_ * HD_);
  cvt(w_o, wo_h, (size_t)D_ * D_);
  cvt(w_gate, wg_h, (size_t)D_ * F_);
  cvt(w_up, wu_h, (size_t)D_ * F_);
  cvt(w_down, wd_h, (size_t)F_ * D_);

  const dim3 blk(256);

  // --- attention half ---
  rmsnorm_f16_kernel<<<dim3((unsigned)MS), blk, 0, stream>>>(x, g_in, xn_h);

  gemm_wmma_kernel<EPI_F16><<<dim3(D_ / 128, (unsigned)(MS / 128)), blk, 0,
                              stream>>>(xn_h, wq_h, q_h, nullptr, nullptr,
                                        nullptr, (int)MS, D_, D_);
  gemm_wmma_kernel<EPI_F16><<<dim3((KVH_ * HD_) / 128, (unsigned)(MS / 128)),
                              blk, 0, stream>>>(xn_h, wk_h, k_h, nullptr,
                                                nullptr, nullptr, (int)MS,
                                                KVH_ * HD_, D_);
  gemm_wmma_kernel<EPI_F16><<<dim3((KVH_ * HD_) / 128, (unsigned)(MS / 128)),
                              blk, 0, stream>>>(xn_h, wv_h, v_h, nullptr,
                                                nullptr, nullptr, (int)MS,
                                                KVH_ * HD_, D_);

  const int qpairs = B_ * S_ * H_ * (HD_ / 2);
  const int kpairs = B_ * S_ * KVH_ * (HD_ / 2);
  rope_f16_kernel<<<dim3((qpairs + 255) / 256), blk, 0, stream>>>(
      q_h, cosb, sinb, H_, qpairs);
  rope_f16_kernel<<<dim3((kpairs + 255) / 256), blk, 0, stream>>>(
      k_h, cosb, sinb, KVH_, kpairs);

  attn_flash_kernel<<<dim3(S_ / 128, H_, B_), blk, 0, stream>>>(q_h, k_h, v_h,
                                                                at_h);

  gemm_wmma_kernel<EPI_RESID_F32><<<dim3(D_ / 128, (unsigned)(MS / 128)), blk,
                                    0, stream>>>(at_h, wo_h, nullptr, x1, x,
                                                 nullptr, (int)MS, D_, D_);

  // --- MLP half ---
  rmsnorm_f16_kernel<<<dim3((unsigned)MS), blk, 0, stream>>>(x1, g_out, xn2_h);

  gemm_wmma_kernel<EPI_SILU_F16><<<dim3(F_ / 128, (unsigned)(MS / 128)), blk,
                                   0, stream>>>(xn2_h, wg_h, g_h, nullptr,
                                                nullptr, nullptr, (int)MS, F_,
                                                D_);
  gemm_wmma_kernel<EPI_MUL_F16><<<dim3(F_ / 128, (unsigned)(MS / 128)), blk, 0,
                                  stream>>>(xn2_h, wu_h, hh, nullptr, nullptr,
                                            g_h, (int)MS, F_, D_);
  gemm_wmma_kernel<EPI_RESID_F32><<<dim3(D_ / 128, (unsigned)(MS / 128)), blk,
                                    0, stream>>>(hh, wd_h, nullptr, out, x1,
                                                 nullptr, (int)MS, D_, F_);
}
